// FlashDenoiseStepWrapper_68951404970058
// MI455X (gfx1250) — matrix operand order repaired and checked by reading, not compiled, not run on hardware
//
#include <hip/hip_runtime.h>
#include <cstdint>
#include <cstddef>

// ---------------- constants ----------------
constexpr int Lc  = 18;
constexpr int Hc  = 1024;
constexpr int NHc = 8;
constexpr int HDc = 256;
constexpr int FFc = 4096;
constexpr int HORc = 50;
constexpr int PREc = 968;
constexpr int Bc  = 8;
constexpr int ADc = 32;
constexpr int Mc  = Bc * HORc;       // 400 token rows
constexpr int Tc  = PREc + HORc;     // 1018 keys
constexpr int TPADc = 1024;          // padded key count
constexpr int KSPLIT = 4;            // attention key-split
constexpr int KCHUNK = TPADc / KSPLIT; // 256 keys per split
constexpr float EPSc = 1e-6f;
constexpr float NEGc = -1e9f;
constexpr float SCALEc = 0.0625f;    // 256^-0.5
constexpr float TWO_PIc = 6.283185307179586f;

typedef __attribute__((ext_vector_type(16))) __bf16 v16bf;
typedef __attribute__((ext_vector_type(8)))  __bf16 v8bf;
typedef __attribute__((ext_vector_type(8)))  float  v8f;
typedef __attribute__((ext_vector_type(4)))  uint32_t v4u;
typedef __attribute__((ext_vector_type(8)))  int v8i_;
typedef __attribute__((ext_vector_type(4)))  int v4i_;

#if defined(__has_builtin)
#if __has_builtin(__builtin_amdgcn_tensor_load_to_lds)
#define HAVE_TDM 1
#endif
#endif

#define DEVINL __device__ __forceinline__

DEVINL v16bf cat16(v8bf lo, v8bf hi) {
  return __builtin_shufflevector(lo, hi, 0,1,2,3,4,5,6,7,8,9,10,11,12,13,14,15);
}
DEVINL v8f wmma_bf16(v16bf a, v16bf b, v8f c) {
  return __builtin_amdgcn_wmma_f32_16x16x32_bf16(false, a, false, b, (short)0, c, false, false);
}
DEVINL float gelu_f(float x) {
  const float kA = 0.7978845608028654f;
  return 0.5f * x * (1.0f + tanhf(kA * (x + 0.044715f * x * x * x)));
}
DEVINL float silu_f(float x) { return x / (1.0f + expf(-x)); }

// ---------------- small VALU kernels ----------------

__global__ __launch_bounds__(256) void k_timeemb(const float* __restrict__ ts,
                                                 float* __restrict__ emb) {
  const int b = blockIdx.x;
  const float t = ts[b];
  for (int j = threadIdx.x; j < Hc; j += blockDim.x) {
    const int f = (j < Hc / 2) ? j : j - Hc / 2;
    const float frac = (float)f / (float)(Hc / 2 - 1);
    const float period = 0.004f * powf(1000.0f, frac);
    const float ang = TWO_PIc / period * t;
    emb[(size_t)b * Hc + j] = (j < Hc / 2) ? sinf(ang) : cosf(ang);
  }
}

__global__ __launch_bounds__(128) void k_gemv_silu(const float* __restrict__ in,
                                                   const float* __restrict__ W,
                                                   const float* __restrict__ bias,
                                                   float* __restrict__ out) {
  const int n = blockIdx.x * 128 + threadIdx.x;
  float acc[Bc] = {};
  for (int k = 0; k < Hc; ++k) {
    const float w = W[(size_t)k * Hc + n];
#pragma unroll
    for (int bb = 0; bb < Bc; ++bb) acc[bb] += in[(size_t)bb * Hc + k] * w;
  }
#pragma unroll
  for (int bb = 0; bb < Bc; ++bb) {
    const float x = acc[bb] + bias[n];
    out[(size_t)bb * Hc + n] = silu_f(x);
  }
}

__global__ __launch_bounds__(128) void k_gemv_nl(const float* __restrict__ cond,
                                                 const float* __restrict__ W,
                                                 float* __restrict__ out) {
  const int l = blockIdx.y;
  const int n = blockIdx.x * 128 + threadIdx.x;
  const float* Wl = W + (size_t)l * Hc * Hc;
  float acc[Bc] = {};
  for (int k = 0; k < Hc; ++k) {
    const float w = Wl[(size_t)k * Hc + n];
#pragma unroll
    for (int bb = 0; bb < Bc; ++bb) acc[bb] += cond[(size_t)bb * Hc + k] * w;
  }
#pragma unroll
  for (int bb = 0; bb < Bc; ++bb)
    out[((size_t)l * Bc + bb) * Hc + n] = acc[bb];
}

__global__ __launch_bounds__(256) void k_actin(const float* __restrict__ x,
                                               const float* __restrict__ w,
                                               const float* __restrict__ bias,
                                               float* __restrict__ h) {
  const int m = blockIdx.x;
  const int n = blockIdx.y * 256 + threadIdx.x;
  float acc = bias[n];
#pragma unroll
  for (int k = 0; k < ADc; ++k) acc += x[(size_t)m * ADc + k] * w[(size_t)k * Hc + n];
  h[(size_t)m * Hc + n] = acc;
}

__global__ __launch_bounds__(128) void k_rope_trig(const unsigned char* __restrict__ mask,
                                                   float* __restrict__ cosb,
                                                   float* __restrict__ sinb) {
  __shared__ int red[128];
  const int b = blockIdx.x, t = blockIdx.y;
  int s = 0;
  for (int i = threadIdx.x; i < PREc; i += 128) s += (int)mask[(size_t)b * PREc + i];
  red[threadIdx.x] = s;
  __syncthreads();
  for (int o = 64; o > 0; o >>= 1) {
    if (threadIdx.x < o) red[threadIdx.x] += red[threadIdx.x + o];
    __syncthreads();
  }
  const float pos = (float)(red[0] + t);
  const int d = threadIdx.x;               // 128 threads, d < 128
  const float invf = powf(10000.0f, -(float)d / 128.0f);
  const float a = pos * invf;
  const size_t o = ((size_t)b * HORc + t) * 128 + d;
  cosb[o] = cosf(a);
  sinb[o] = sinf(a);
}

__global__ __launch_bounds__(256) void k_rmsnorm(const float* __restrict__ x,
                                                 const float* __restrict__ nw,
                                                 const float* __restrict__ sv,
                                                 __bf16* __restrict__ out) {
  __shared__ float red[256];
  const int m = blockIdx.x;
  const int b = m / HORc;
  const float* xr = x + (size_t)m * Hc;
  float s = 0.f;
  for (int i = threadIdx.x; i < Hc; i += 256) { const float v = xr[i]; s += v * v; }
  red[threadIdx.x] = s;
  __syncthreads();
  for (int o = 128; o > 0; o >>= 1) {
    if (threadIdx.x < o) red[threadIdx.x] += red[threadIdx.x + o];
    __syncthreads();
  }
  const float r = rsqrtf(red[0] / (float)Hc + EPSc);
  for (int i = threadIdx.x; i < Hc; i += 256) {
    const float v = xr[i] * r * (1.0f + nw[i]) * (1.0f + sv[(size_t)b * Hc + i]);
    out[(size_t)m * Hc + i] = (__bf16)v;
  }
}

// transpose + convert: W[K][N] f32 -> Wt[N][K] bf16 (all dims multiples of 32)
__global__ __launch_bounds__(256) void k_wt(const float* __restrict__ W,
                                            __bf16* __restrict__ Wt,
                                            int K, int N) {
  __shared__ float tile[32][33];
  const int k0 = blockIdx.y * 32, n0 = blockIdx.x * 32;
  const int tx = threadIdx.x & 31, ty = threadIdx.x >> 5;   // 8 rows of 32
  for (int r = ty; r < 32; r += 8)
    tile[r][tx] = W[(size_t)(k0 + r) * N + n0 + tx];
  __syncthreads();
  for (int r = ty; r < 32; r += 8)
    Wt[(size_t)(n0 + r) * K + k0 + tx] = (__bf16)tile[tx][r];
}

// ---------------- A-tile staging into LDS (two CDNA5 async paths) ----------------

// per-lane async copies, tracked by ASYNCcnt
template <int KK>
DEVINL void stage_async(const __bf16* __restrict__ A, int m0, int tid) {
  for (int idx = tid; idx < 2 * KK; idx += 256) {
    const int row = idx / (KK / 8);
    const int c = idx % (KK / 8);
    const __bf16* gp = A + (size_t)(m0 + row) * KK + c * 8;
    const uint64_t ga = (uint64_t)(uintptr_t)gp;
    const uint32_t la = (uint32_t)idx * 16u;   // row*2*KK + c*16 bytes
    asm volatile("global_load_async_to_lds_b128 %0, %1, off"
                 :: "v"(la), "v"(ga) : "memory");
  }
  asm volatile("s_wait_asynccnt 0x0" ::: "memory");
}

#ifdef HAVE_TDM
// single Tensor-Data-Mover descriptor: 2-D tile of 16 rows x KK bf16, row
// stride KK elements, into LDS offset 0.  D# per CDNA5 ISA.
// This toolchain declares the 6-arg builtin:
//   (uint32x4 g0, int32x8 g1, int32x4 g2, int32x4 g3, int32x8, i32 cpol)
template <int KK>
DEVINL void stage_tdm(const __bf16* __restrict__ A, int m0) {
  const uint64_t ga = (uint64_t)(uintptr_t)(A + (size_t)m0 * KK);
  const uint32_t kk = (uint32_t)KK;
  v4u g0;
  g0[0] = 1u;                                        // count=1, no gather
  g0[1] = 0u;                                        // lds_addr = 0
  g0[2] = (uint32_t)ga;                              // global_addr[31:0]
  g0[3] = (uint32_t)((ga >> 32) & 0x01FFFFFFu) | 0x80000000u;  // addr[56:32] | type=2
  v8i_ g1;
  g1[0] = (int)(1u << 16);                           // wg_mask=0, data_size=1 (2B)
  g1[1] = (int)((kk & 0xFFFFu) << 16);               // tensor_dim0 lo16
  g1[2] = (int)(((kk >> 16) & 0xFFFFu) | (16u << 16)); // tensor_dim0 hi16 | tensor_dim1 lo16=16
  g1[3] = (int)(kk << 16);                           // tensor_dim1 hi16=0 | tile_dim0=KK
  g1[4] = 16;                                        // tile_dim1=16, tile_dim2=0
  g1[5] = (int)kk;                                   // tensor_dim0_stride lo32
  g1[6] = 0;                                         // stride hi16=0, dim1_stride lo16=0
  g1[7] = 0;
  v4i_ gz4 = {0, 0, 0, 0};
  v8i_ gz8 = {0, 0, 0, 0, 0, 0, 0, 0};
  __builtin_amdgcn_tensor_load_to_lds(g0, g1, gz4, gz4, gz8, 0);
  __builtin_amdgcn_s_wait_tensorcnt((short)0);
}
#endif

// ---------------- WMMA GEMM (bf16 x bf16 -> f32) ----------------
// A: [M,KK] bf16 row-major.  Wt/W2t: [N,KK] bf16 (pre-transposed).
// One wave computes one 16x16 C tile; 8 waves/block share the M-tile.
// If KK <= 2048 the shared A tile (16 x KK bf16, <=64KB) is staged into LDS
// once per block: via the TDM (TENSORcnt) in MODE 0, via
// global_load_async_to_lds_b128 (ASYNCcnt) otherwise.  Depth-2 software
// pipeline on the weight stream.
// MODE 0: outf = acc
// MODE 1: outf = resid + gatev[b][n]*acc          (N == H)
// MODE 2: outh = gelu(acc)*acc2  (dual-B, bf16 out)
template <int MODE, int KK>
__global__ __launch_bounds__(256) void k_gemm(const __bf16* __restrict__ A,
                                              const __bf16* __restrict__ Wt,
                                              const __bf16* __restrict__ W2t,
                                              float* __restrict__ outf,
                                              __bf16* __restrict__ outh,
                                              const float* __restrict__ resid,
                                              const float* __restrict__ gatev,
                                              int N) {
  constexpr bool STAGE = (KK <= 2048);
  extern __shared__ __bf16 Alds[];

  const int tid = threadIdx.x;
  const int wid = tid >> 5, lane = tid & 31;
  const int half = lane >> 4, l15 = lane & 15;
  const int n0 = (blockIdx.x * 8 + wid) * 16;
  const int m0 = blockIdx.y * 16;

  if constexpr (STAGE) {
#ifdef HAVE_TDM
    if constexpr (MODE == 0) {
      if (wid == 0) stage_tdm<KK>(A, m0);
    } else {
      stage_async<KK>(A, m0, tid);
    }
#else
    stage_async<KK>(A, m0, tid);
#endif
    __syncthreads();
  }

  const __bf16* Ag = A + (size_t)(m0 + l15) * KK + 8 * half;        // !STAGE path
  const __bf16* Wr = Wt + (size_t)(n0 + l15) * KK + 8 * half;
  const __bf16* W2r = (MODE == 2) ? (W2t + (size_t)(n0 + l15) * KK + 8 * half) : nullptr;

  auto loadA = [&](int k0) -> v16bf {
    if constexpr (STAGE) {
      const __bf16* p = &Alds[(size_t)l15 * KK + 8 * half + k0];
      return cat16(*(const v8bf*)p, *(const v8bf*)(p + 16));
    } else {
      return cat16(*(const v8bf*)(Ag + k0), *(const v8bf*)(Ag + k0 + 16));
    }
  };
  auto loadB = [&](int k0) -> v16bf {
    return cat16(*(const v8bf*)(Wr + k0), *(const v8bf*)(Wr + k0 + 16));
  };
  auto loadB2 = [&](int k0) -> v16bf {
    return cat16(*(const v8bf*)(W2r + k0), *(const v8bf*)(W2r + k0 + 16));
  };

  v8f acc = {};
  v8f acc2 = {};
  // depth-2 software pipeline on operand streams
  v16bf aP0 = loadA(0),  aP1 = loadA(32);
  v16bf bP0 = loadB(0),  bP1 = loadB(32);
  v16bf c0, c1;
  if constexpr (MODE == 2) { c0 = loadB2(0); c1 = loadB2(32); }
  int k0 = 0;
  for (; k0 + 64 < KK; k0 += 32) {
    __builtin_prefetch(Wr + k0 + 1024, 0, 0);
    v16bf an = loadA(k0 + 64);
    v16bf bn = loadB(k0 + 64);
    v16bf cn;
    if constexpr (MODE == 2) cn = loadB2(k0 + 64);
    acc = wmma_bf16(aP0, bP0, acc);
    if constexpr (MODE == 2) acc2 = wmma_bf16(aP0, c0, acc2);
    aP0 = aP1; bP0 = bP1; aP1 = an; bP1 = bn;
    if constexpr (MODE == 2) { c0 = c1; c1 = cn; }
  }
  acc = wmma_bf16(aP0, bP0, acc);
  if constexpr (MODE == 2) acc2 = wmma_bf16(aP0, c0, acc2);
  acc = wmma_bf16(aP1, bP1, acc);
  if constexpr (MODE == 2) acc2 = wmma_bf16(aP1, c1, acc2);

#pragma unroll
  for (int r = 0; r < 8; ++r) {
    const int m = m0 + r + 8 * half;
    const int n = n0 + l15;
    const size_t idx = (size_t)m * N + n;
    if (MODE == 0) {
      outf[idx] = acc[r];
    } else if (MODE == 1) {
      const int b = m / HORc;
      outf[idx] = resid[idx] + gatev[(size_t)b * N + n] * acc[r];
    } else {
      outh[idx] = (__bf16)(gelu_f(acc[r]) * acc2[r]);
    }
  }
}

// ---------------- RoPE apply ----------------
__global__ __launch_bounds__(256) void k_rope_apply(const float* __restrict__ qin,
                                                    const float* __restrict__ kin,
                                                    const float* __restrict__ cosb,
                                                    const float* __restrict__ sinb,
                                                    __bf16* __restrict__ qout,
                                                    float* __restrict__ kout) {
  const int m = blockIdx.x;                 // b*50 + t
  const int b = m / HORc, t = m % HORc;
  const float* cb = cosb + ((size_t)b * HORc + t) * 128;
  const float* sb = sinb + ((size_t)b * HORc + t) * 128;
  if (blockIdx.y == 0) {
    for (int i = threadIdx.x; i < NHc * HDc; i += blockDim.x) {
      const int hh = i >> 8, d = i & 255;
      const float x = qin[(size_t)m * (NHc * HDc) + i];
      const float other = qin[(size_t)m * (NHc * HDc) + hh * HDc + ((d < 128) ? d + 128 : d - 128)];
      const float rot = (d < 128) ? -other : other;
      const float o = x * cb[d & 127] + rot * sb[d & 127];
      qout[((size_t)(b * 400 + hh * HORc + t)) * HDc + d] = (__bf16)o;
    }
  } else {
    for (int d = threadIdx.x; d < HDc; d += blockDim.x) {
      const float x = kin[(size_t)m * HDc + d];
      const float other = kin[(size_t)m * HDc + ((d < 128) ? d + 128 : d - 128)];
      const float rot = (d < 128) ? -other : other;
      kout[(size_t)m * HDc + d] = x * cb[d & 127] + rot * sb[d & 127];
    }
  }
}

// ---------------- per-layer K/V staging to bf16 ----------------
__global__ __launch_bounds__(256) void k_cvt_keys(const float* __restrict__ pk,
                                                  const float* __restrict__ krope,
                                                  __bf16* __restrict__ kb16,
                                                  int layer) {
  const int b = blockIdx.x, key = blockIdx.y;
  const int hd = threadIdx.x;
  float v = 0.f;
  if (key < PREc)
    v = pk[(((size_t)b * Lc + layer) * PREc + key) * HDc + hd];
  else if (key < Tc)
    v = krope[((size_t)b * HORc + (key - PREc)) * HDc + hd];
  kb16[((size_t)b * TPADc + key) * HDc + hd] = (__bf16)v;
}

// V transpose via 32x32 LDS tiles: coalesced f32 reads, coalesced bf16 writes
__global__ __launch_bounds__(256) void k_cvt_vt(const float* __restrict__ pv,
                                                const float* __restrict__ vbuf,
                                                __bf16* __restrict__ vt,
                                                int layer) {
  __shared__ float tile[32][33];
  const int b = blockIdx.x;
  const int key0 = blockIdx.y * 32;        // 32 key rows
  const int hd0 = blockIdx.z * 32;         // 32 hd cols
  const int tx = threadIdx.x & 31, ty = threadIdx.x >> 5;
  for (int r = ty; r < 32; r += 8) {
    const int key = key0 + r;
    float v = 0.f;
    if (key < PREc)
      v = pv[(((size_t)b * Lc + layer) * PREc + key) * HDc + hd0 + tx];
    else if (key < Tc)
      v = vbuf[((size_t)b * HORc + (key - PREc)) * HDc + hd0 + tx];
    tile[r][tx] = v;
  }
  __syncthreads();
  for (int r = ty; r < 32; r += 8)
    vt[((size_t)b * HDc + hd0 + r) * TPADc + key0 + tx] = (__bf16)tile[tx][r];
}

// ---------------- WMMA flash attention (key-split) ----------------
__global__ __launch_bounds__(128) void k_attn(const __bf16* __restrict__ qb,   // (b*400+q)*256
                                              const __bf16* __restrict__ kb16, // [B][1024][256]
                                              const __bf16* __restrict__ vt,   // [B][256][1024]
                                              float* __restrict__ Opart,       // [Z][B*400][256]
                                              float* __restrict__ mstat,       // [Z][B*400]
                                              float* __restrict__ lstat) {     // [Z][B*400]
  __shared__ float plds[4][16][34];
  const int wid = threadIdx.x >> 5;
  const int lane = threadIdx.x & 31;
  const int half = lane >> 4, l15 = lane & 15;
  const int b = blockIdx.x;
  const int z = blockIdx.z;
  int qtile = blockIdx.y * 4 + wid;
  const bool active = qtile < (Mc / 16);
  if (!active) qtile = 0;
  const int q0 = qtile * 16;

  const __bf16* qrow = qb + ((size_t)(b * 400 + q0 + l15)) * HDc + 16 * half;
  v16bf Qa[8];
#pragma unroll
  for (int c = 0; c < 8; ++c) {
    v8bf lo = *(const v8bf*)(qrow + c * 32);
    v8bf hi = *(const v8bf*)(qrow + c * 32 + 8);
    Qa[c] = cat16(lo, hi);
  }

  v8f O[16];
  {
    v8f zz = {};
#pragma unroll
    for (int tt = 0; tt < 16; ++tt) O[tt] = zz;
  }
  float mrow[8], lrow[8];
#pragma unroll
  for (int r = 0; r < 8; ++r) { mrow[r] = -1e30f; lrow[r] = 0.f; }

  const __bf16* kbb = kb16 + (size_t)b * TPADc * HDc;
  const __bf16* vtb = vt + (size_t)b * HDc * TPADc;

  const int kbeg = z * KCHUNK;
  const int kend = kbeg + KCHUNK;
  for (int kb0 = kbeg; kb0 < kend; kb0 += 32) {
    // ---- S = Q @ K^T for two 16-key sub-tiles (pipelined b128 loads) ----
    v8f s[2];
#pragma unroll
    for (int sub = 0; sub < 2; ++sub) {
      const int key = kb0 + sub * 16 + l15;            // < 1024 always (padded)
      const __bf16* kr = kbb + (size_t)key * HDc + 16 * half;
      v8bf lo = *(const v8bf*)(kr);
      v8bf hi = *(const v8bf*)(kr + 8);
      v8f sv = {};
#pragma unroll
      for (int c = 0; c < 7; ++c) {
        v8bf lo_n = *(const v8bf*)(kr + (c + 1) * 32);
        v8bf hi_n = *(const v8bf*)(kr + (c + 1) * 32 + 8);
        sv = wmma_bf16(Qa[c], cat16(lo, hi), sv);
        lo = lo_n; hi = hi_n;
      }
      sv = wmma_bf16(Qa[7], cat16(lo, hi), sv);
      s[sub] = sv;
    }
    // ---- mask + online softmax (row stats via width-16 shfl_xor) ----
    float facr[8];
#pragma unroll
    for (int r = 0; r < 8; ++r) {
      const int qg = q0 + r + 8 * half;
      const int tq = qg % HORc;
      const int key0 = kb0 + l15;
      const int key1 = kb0 + 16 + l15;
      float v0 = s[0][r] * SCALEc;
      float v1 = s[1][r] * SCALEc;
      if (key0 >= Tc || (key0 >= PREc && key0 - PREc > tq)) v0 = NEGc;
      if (key1 >= Tc || (key1 >= PREc && key1 - PREc > tq)) v1 = NEGc;
      float rm = fmaxf(v0, v1);
      rm = fmaxf(rm, __shfl_xor(rm, 1, 16));
      rm = fmaxf(rm, __shfl_xor(rm, 2, 16));
      rm = fmaxf(rm, __shfl_xor(rm, 4, 16));
      rm = fmaxf(rm, __shfl_xor(rm, 8, 16));
      const float mn = fmaxf(mrow[r], rm);
      const float fac = __expf(mrow[r] - mn);
      const float p0 = __expf(v0 - mn);
      const float p1 = __expf(v1 - mn);
      float rs = p0 + p1;
      rs += __shfl_xor(rs, 1, 16);
      rs += __shfl_xor(rs, 2, 16);
      rs += __shfl_xor(rs, 4, 16);
      rs += __shfl_xor(rs, 8, 16);
      lrow[r] = lrow[r] * fac + rs;
      mrow[r] = mn;
      facr[r] = fac;
      plds[wid][r + 8 * half][l15]      = p0;   // P in S-layout -> LDS
      plds[wid][r + 8 * half][16 + l15] = p1;
    }
#pragma unroll
    for (int tt = 0; tt < 16; ++tt)
#pragma unroll
      for (int r = 0; r < 8; ++r) O[tt][r] *= facr[r];
    // read P back in A-operand layout (wave-local LDS transpose)
    v16bf Pa;
#pragma unroll
    for (int e = 0; e < 8; ++e) {
      Pa[e]     = (__bf16)plds[wid][l15][16 * half + e];
      Pa[8 + e] = (__bf16)plds[wid][l15][16 * half + 8 + e];
    }
    // ---- O += P @ V over 16 hd tiles (pipelined b128 loads) ----
    {
      const __bf16* vr = vtb + (size_t)l15 * TPADc + kb0 + 16 * half;
      v8bf vlo = *(const v8bf*)(vr);
      v8bf vhi = *(const v8bf*)(vr + 8);
#pragma unroll
      for (int tt = 0; tt < 15; ++tt) {
        const __bf16* vrn = vtb + (size_t)((tt + 1) * 16 + l15) * TPADc + kb0 + 16 * half;
        v8bf nlo = *(const v8bf*)(vrn);
        v8bf nhi = *(const v8bf*)(vrn + 8);
        O[tt] = wmma_bf16(Pa, cat16(vlo, vhi), O[tt]);
        vlo = nlo; vhi = nhi;
      }
      O[15] = wmma_bf16(Pa, cat16(vlo, vhi), O[15]);
    }
  }

  if (active) {
#pragma unroll
    for (int r = 0; r < 8; ++r) {
      const size_t rowIdx = ((size_t)z * Bc + b) * 400 + q0 + r + 8 * half;
      if (l15 == 0) { mstat[rowIdx] = mrow[r]; lstat[rowIdx] = lrow[r]; }
    }
#pragma unroll
    for (int tt = 0; tt < 16; ++tt)
#pragma unroll
      for (int r = 0; r < 8; ++r) {
        const size_t rowIdx = ((size_t)z * Bc + b) * 400 + q0 + r + 8 * half;
        Opart[rowIdx * HDc + tt * 16 + l15] = O[tt][r];
      }
  }
}

// merge KSPLIT partials -> att bf16 in WO-GEMM layout [400][2048]
__global__ __launch_bounds__(256) void k_attn_merge(const float* __restrict__ Opart,
                                                    const float* __restrict__ mstat,
                                                    const float* __restrict__ lstat,
                                                    __bf16* __restrict__ att) {
  const int row = blockIdx.x;               // b*400 + q,  q = h*50 + t
  const int b = row / 400, q = row % 400;
  const int hh = q / HORc, t = q % HORc;
  float mv[KSPLIT], lv[KSPLIT];
  float mx = -1e30f;
#pragma unroll
  for (int zz = 0; zz < KSPLIT; ++zz) {
    mv[zz] = mstat[((size_t)zz * Bc + b) * 400 + q];
    lv[zz] = lstat[((size_t)zz * Bc + b) * 400 + q];
    mx = fmaxf(mx, mv[zz]);
  }
  float w[KSPLIT], ltot = 0.f;
#pragma unroll
  for (int zz = 0; zz < KSPLIT; ++zz) {
    w[zz] = __expf(mv[zz] - mx);
    ltot += w[zz] * lv[zz];
  }
  const float inv = 1.0f / ltot;
  const int d = threadIdx.x;                // HD == 256 == blockDim
  float s = 0.f;
#pragma unroll
  for (int zz = 0; zz < KSPLIT; ++zz)
    s += w[zz] * Opart[(((size_t)zz * Bc + b) * 400 + q) * HDc + d];
  att[((size_t)(b * HORc + t)) * (NHc * HDc) + hh * HDc + d] = (__bf16)(s * inv);
}

// ---------------- final projection ----------------
__global__ __launch_bounds__(32) void k_final(const __bf16* __restrict__ nf,
                                              const float* __restrict__ w,
                                              const float* __restrict__ bias,
                                              float* __restrict__ out) {
  const int m = blockIdx.x;
  const int j = threadIdx.x;
  float acc = bias[j];
  for (int k = 0; k < Hc; ++k)
    acc += (float)nf[(size_t)m * Hc + k] * w[(size_t)k * ADc + j];
  out[(size_t)m * ADc + j] = acc;
}

// ---------------- host orchestration ----------------
extern "C" void kernel_launch(void* const* d_in, const int* in_sizes, int n_in,
                              void* d_out, int out_size, void* d_ws, size_t ws_size,
                              hipStream_t stream) {
  (void)in_sizes; (void)n_in; (void)out_size; (void)ws_size;
  const float* pk    = (const float*)d_in[0];
  const float* pv    = (const float*)d_in[1];
  const float* x_t   = (const float*)d_in[2];
  const float* tstep = (const float*)d_in[3];
  const unsigned char* pmask = (const unsigned char*)d_in[4];
  const float* aiw = (const float*)d_in[5];
  const float* aib = (const float*)d_in[6];
  const float* aow = (const float*)d_in[7];
  const float* aob = (const float*)d_in[8];
  const float* tiw = (const float*)d_in[9];
  const float* tib = (const float*)d_in[10];
  const float* tow = (const float*)d_in[11];
  const float* tob = (const float*)d_in[12];
  const float* wq = (const float*)d_in[13];
  const float* wk = (const float*)d_in[14];
  const float* wv = (const float*)d_in[15];
  const float* wo = (const float*)d_in[16];
  const float* wg = (const float*)d_in[17];
  const float* wu = (const float*)d_in[18];
  const float* wd = (const float*)d_in[19];
  const float* innw = (const float*)d_in[20];
  const float* insw = (const float*)d_in[21];
  const float* ingw = (const float*)d_in[22];
  const float* pnw = (const float*)d_in[23];
  const float* psw = (const float*)d_in[24];
  const float* pgw = (const float*)d_in[25];
  const float* fnw = (const float*)d_in[26];
  const float* fsw = (const float*)d_in[27];

  char* p = (char*)d_ws;
  auto alloc = [&](size_t bytes) -> char* {
    char* r = p;
    p += (bytes + 255) & ~(size_t)255;
    return r;
  };
  float* emb  = (float*)alloc((size_t)Bc * Hc * 4);
  float* t1   = (float*)alloc((size_t)Bc * Hc * 4);
  float* cond = (float*)alloc((size_t)Bc * Hc * 4);
  float* ada_is = (float*)alloc((size_t)Lc * Bc * Hc * 4);
  float* ada_ig = (float*)alloc((size_t)Lc * Bc * Hc * 4);
  float* ada_ps = (float*)alloc((size_t)Lc * Bc * Hc * 4);
  float* ada_pg = (float*)alloc((size_t)Lc * Bc * Hc * 4);
  float* fvec = (float*)alloc((size_t)Bc * Hc * 4);
  float* cosb = (float*)alloc((size_t)Bc * HORc * 128 * 4);
  float* sinb = (float*)alloc((size_t)Bc * HORc * 128 * 4);
  float* h    = (float*)alloc((size_t)Mc * Hc * 4);
  float* h1   = (float*)alloc((size_t)Mc * Hc * 4);
  float* qbuf = (float*)alloc((size_t)Mc * NHc * HDc * 4);
  float* kbuf = (float*)alloc((size_t)Mc * HDc * 4);
  float* vbuf = (float*)alloc((size_t)Mc * HDc * 4);
  float* krope = (float*)alloc((size_t)Mc * HDc * 4);
  float* Opart = (float*)alloc((size_t)KSPLIT * Bc * 400 * HDc * 4);
  float* mstat = (float*)alloc((size_t)KSPLIT * Bc * 400 * 4);
  float* lstat = (float*)alloc((size_t)KSPLIT * Bc * 400 * 4);
  __bf16* nrm   = (__bf16*)alloc((size_t)Mc * Hc * 2);
  __bf16* qrope = (__bf16*)alloc((size_t)Bc * 400 * HDc * 2);
  __bf16* attb  = (__bf16*)alloc((size_t)Mc * NHc * HDc * 2);
  __bf16* actb  = (__bf16*)alloc((size_t)Mc * FFc * 2);
  __bf16* kb16  = (__bf16*)alloc((size_t)Bc * TPADc * HDc * 2);
  __bf16* vt    = (__bf16*)alloc((size_t)Bc * HDc * TPADc * 2);
  __bf16* wqT = (__bf16*)alloc((size_t)(NHc * HDc) * Hc * 2);
  __bf16* wkT = (__bf16*)alloc((size_t)HDc * Hc * 2);
  __bf16* wvT = (__bf16*)alloc((size_t)HDc * Hc * 2);
  __bf16* woT = (__bf16*)alloc((size_t)Hc * (NHc * HDc) * 2);
  __bf16* wgT = (__bf16*)alloc((size_t)FFc * Hc * 2);
  __bf16* wuT = (__bf16*)alloc((size_t)FFc * Hc * 2);
  __bf16* wdT = (__bf16*)alloc((size_t)Hc * FFc * 2);

  // ---- preamble ----
  k_timeemb<<<Bc, 256, 0, stream>>>(tstep, emb);
  k_gemv_silu<<<Hc / 128, 128, 0, stream>>>(emb, tiw, tib, t1);
  k_gemv_silu<<<Hc / 128, 128, 0, stream>>>(t1, tow, tob, cond);
  {
    dim3 g(Hc / 128, Lc);
    k_gemv_nl<<<g, 128, 0, stream>>>(cond, insw, ada_is);
    k_gemv_nl<<<g, 128, 0, stream>>>(cond, ingw, ada_ig);
    k_gemv_nl<<<g, 128, 0, stream>>>(cond, psw, ada_ps);
    k_gemv_nl<<<g, 128, 0, stream>>>(cond, pgw, ada_pg);
    dim3 g1(Hc / 128, 1);
    k_gemv_nl<<<g1, 128, 0, stream>>>(cond, fsw, fvec);
  }
  k_actin<<<dim3(Mc, Hc / 256), 256, 0, stream>>>(x_t, aiw, aib, h);
  k_rope_trig<<<dim3(Bc, HORc), 128, 0, stream>>>(pmask, cosb, sinb);

  // ---- layers ----
  for (int i = 0; i < Lc; ++i) {
    const float* wq_l = wq + (size_t)i * Hc * NHc * HDc;
    const float* wk_l = wk + (size_t)i * Hc * HDc;
    const float* wv_l = wv + (size_t)i * Hc * HDc;
    const float* wo_l = wo + (size_t)i * NHc * HDc * Hc;
    const float* wg_l = wg + (size_t)i * Hc * FFc;
    const float* wu_l = wu + (size_t)i * Hc * FFc;
    const float* wd_l = wd + (size_t)i * FFc * Hc;

    // layer prologue: transpose+convert weights to bf16 [N,K]
    k_wt<<<dim3((NHc * HDc) / 32, Hc / 32), 256, 0, stream>>>(wq_l, wqT, Hc, NHc * HDc);
    k_wt<<<dim3(HDc / 32, Hc / 32), 256, 0, stream>>>(wk_l, wkT, Hc, HDc);
    k_wt<<<dim3(HDc / 32, Hc / 32), 256, 0, stream>>>(wv_l, wvT, Hc, HDc);
    k_wt<<<dim3(Hc / 32, (NHc * HDc) / 32), 256, 0, stream>>>(wo_l, woT, NHc * HDc, Hc);
    k_wt<<<dim3(FFc / 32, Hc / 32), 256, 0, stream>>>(wg_l, wgT, Hc, FFc);
    k_wt<<<dim3(FFc / 32, Hc / 32), 256, 0, stream>>>(wu_l, wuT, Hc, FFc);
    k_wt<<<dim3(Hc / 32, FFc / 32), 256, 0, stream>>>(wd_l, wdT, FFc, Hc);

    // attention block
    k_rmsnorm<<<Mc, 256, 0, stream>>>(h, innw + (size_t)i * Hc, ada_is + (size_t)i * Bc * Hc, nrm);
    k_gemm<0, Hc><<<dim3((NHc * HDc) / 128, Mc / 16), 256, Hc * 32, stream>>>(
        nrm, wqT, nullptr, qbuf, nullptr, nullptr, nullptr, NHc * HDc);
    k_gemm<0, Hc><<<dim3(HDc / 128, Mc / 16), 256, Hc * 32, stream>>>(
        nrm, wkT, nullptr, kbuf, nullptr, nullptr, nullptr, HDc);
    k_gemm<0, Hc><<<dim3(HDc / 128, Mc / 16), 256, Hc * 32, stream>>>(
        nrm, wvT, nullptr, vbuf, nullptr, nullptr, nullptr, HDc);
    k_rope_apply<<<dim3(Mc, 2), 256, 0, stream>>>(qbuf, kbuf, cosb, sinb, qrope, krope);
    k_cvt_keys<<<dim3(Bc, TPADc), 256, 0, stream>>>(pk, krope, kb16, i);
    k_cvt_vt<<<dim3(Bc, TPADc / 32, HDc / 32), 256, 0, stream>>>(pv, vbuf, vt, i);
    k_attn<<<dim3(Bc, 7, KSPLIT), 128, 0, stream>>>(qrope, kb16, vt, Opart, mstat, lstat);
    k_attn_merge<<<Bc * 400, HDc, 0, stream>>>(Opart, mstat, lstat, attb);
    k_gemm<1, NHc * HDc><<<dim3(Hc / 128, Mc / 16), 256, (NHc * HDc) * 32, stream>>>(
        attb, woT, nullptr, h1, nullptr, h, ada_ig + (size_t)i * Bc * Hc, Hc);

    // MLP block
    k_rmsnorm<<<Mc, 256, 0, stream>>>(h1, pnw + (size_t)i * Hc, ada_ps + (size_t)i * Bc * Hc, nrm);
    k_gemm<2, Hc><<<dim3(FFc / 128, Mc / 16), 256, Hc * 32, stream>>>(
        nrm, wgT, wuT, nullptr, actb, nullptr, nullptr, FFc);
    k_gemm<1, FFc><<<dim3(Hc / 128, Mc / 16), 256, 0, stream>>>(
        actb, wdT, nullptr, h, nullptr, h1, ada_pg + (size_t)i * Bc * Hc, Hc);
  }

  // ---- epilogue ----
  k_rmsnorm<<<Mc, 256, 0, stream>>>(h, fnw, fvec, nrm);
  k_final<<<Mc, 32, 0, stream>>>(nrm, aow, aob, (float*)d_out);
}
